// RiskyObject_4758823764534
// MI455X (gfx1250) — compile-verified, matrix operand order checked by reading, not run on hardware
//
#include <hip/hip_runtime.h>
#include <hip/hip_bf16.h>

// ---------------------------------------------------------------------------
// CDNA5 (gfx1250) implementation. wave32, WMMA bf16 16x16x32, f32 accumulate.
// Phase 1 (parallel): XV = relu(flow@Wphi^T), GI0 = x_t@Wih0^T, GIc (tiny).
// Phase 2 (serial):  one persistent 512-thread workgroup, state in LDS,
//                    WMMA for all 30x256x768-class matmuls, TDM staging of
//                    the per-step GIc block into LDS (overlapped with WMMA).
// ---------------------------------------------------------------------------

typedef __attribute__((ext_vector_type(8)))  __bf16 v8bf;
typedef __attribute__((ext_vector_type(16))) __bf16 v16bf;
typedef __attribute__((ext_vector_type(8)))  float  v8f;
typedef __attribute__((ext_vector_type(4)))  unsigned int v4u;
typedef __attribute__((ext_vector_type(8)))  int v8i;
typedef __attribute__((ext_vector_type(4)))  int v4i;

#define TSTEPS 500
#define NBX    30
#define HD     256
#define HCD    32
#define XD     2048

// TDM builtin arity differs between toolchains (probe-verified): therock
// headers ship amd_gfx1250_TDM.h and use the 6-arg form; ROCm 7.2 is 5-arg.
#if defined(__has_include)
#if __has_include(<hip/amd_detail/amd_gfx1250_TDM.h>)
#define TDM_SIX_ARGS 1
#else
#define TDM_SIX_ARGS 0
#endif
#else
#define TDM_SIX_ARGS 0
#endif

// ---- WMMA fragment helpers (ISA 16-bit A/B layout, 16x16x32) --------------
__device__ __forceinline__ v16bf ld_frag(const __bf16* p) {
  v8bf lo = *(const v8bf*)p;        // k = kA .. kA+7
  v8bf hi = *(const v8bf*)(p + 16); // k = kA+16 .. kA+23
  return __builtin_shufflevector(lo, hi, 0,1,2,3,4,5,6,7,8,9,10,11,12,13,14,15);
}

__device__ __forceinline__ v8f wmma_bf16(v16bf a, v16bf b, v8f c) {
  return __builtin_amdgcn_wmma_f32_16x16x32_bf16(false, a, false, b, (short)0, c,
                                                 false, false);
}

__device__ __forceinline__ float sigm(float x) { return 1.f / (1.f + __expf(-x)); }

// ---- TDM: 1-D contiguous fp32 copy global -> LDS (D# per ISA ch.8) --------
__device__ __forceinline__ void tdm_load_1d_f32(const float* gsrc, unsigned lds_off,
                                                unsigned nelem) {
  unsigned long ga = (unsigned long)(size_t)(const void*)gsrc;
  v4u g0;
  g0[0] = 1u;                                          // count=1 (valid D#)
  g0[1] = lds_off;                                     // lds_addr (bytes)
  g0[2] = (unsigned)(ga & 0xffffffffu);                // global_addr[31:0]
  g0[3] = (unsigned)((ga >> 32) & 0x01ffffffu)         // global_addr[56:32]
        | (2u << 30);                                  // type = 2 ("image")
  v8i g1;
  g1[0] = (int)(2u << 16);                             // data_size = 4 bytes
  g1[1] = (int)((nelem & 0xffffu) << 16);              // tensor_dim0[15:0]
  g1[2] = (int)((nelem >> 16) | (1u << 16));           // dim0[31:16], dim1=1
  g1[3] = (int)((nelem & 0xffffu) << 16);              // tile_dim0 = nelem
  g1[4] = 1;                                           // tile_dim1 = 1
  g1[5] = (int)nelem;                                  // dim0_stride
  g1[6] = 0; g1[7] = 0;
  v4i z4 = {0, 0, 0, 0};
#if TDM_SIX_ARGS
  v8i z8 = {0, 0, 0, 0, 0, 0, 0, 0};
  __builtin_amdgcn_tensor_load_to_lds(g0, g1, z4, z4, z8, 0);
#else
  __builtin_amdgcn_tensor_load_to_lds(g0, g1, z4, z4, 0);
#endif
}

// ---------------------------------------------------------------------------
// k_f2bf: fp32 -> bf16 weight conversion
// ---------------------------------------------------------------------------
__global__ void k_f2bf(const float* __restrict__ src, __bf16* __restrict__ dst, int n) {
  int i = blockIdx.x * 256 + threadIdx.x;
  if (i < n) dst[i] = (__bf16)src[i];
}

// ---------------------------------------------------------------------------
// k_xv: XV[15500,256] = relu(flow[15500,2048] @ WphiB^T + bphi), bf16 out
// block = one 16-row M-block; 8 waves x 2 N-tiles each (A-fragment reuse)
// ---------------------------------------------------------------------------
__global__ void k_xv(const float* __restrict__ flow, const __bf16* __restrict__ WphiB,
                     const float* __restrict__ bphi, __bf16* __restrict__ XV) {
  const int lane = threadIdx.x & 31;
  const int wid  = threadIdx.x >> 5;                  // N-pair id, 0..7
  const int tm   = blockIdx.x;                        // 969 M-blocks
  const int r    = lane & 15;
  const int kA   = (lane & 16) ? 8 : 0;
  int arow = tm * 16 + r; if (arow > 15499) arow = 15499;      // clamp pad rows
  const float*  ap  = flow  + (long)arow * XD + kA;
  const int     n0  = wid * 32;                       // first of two N-tiles
  const __bf16* bp0 = WphiB + (long)(n0 + r) * XD + kA;
  const __bf16* bp1 = WphiB + (long)(n0 + 16 + r) * XD + kA;
  v8f acc0 = {}, acc1 = {};
  for (int kk = 0; kk < XD; kk += 32) {
    v16bf a;
#pragma unroll
    for (int i = 0; i < 8; ++i) { a[i] = (__bf16)ap[kk + i]; a[i + 8] = (__bf16)ap[kk + 16 + i]; }
    acc0 = wmma_bf16(a, ld_frag(bp0 + kk), acc0);
    acc1 = wmma_bf16(a, ld_frag(bp1 + kk), acc1);
  }
  const int rbase = tm * 16 + ((lane & 16) ? 8 : 0);
#pragma unroll
  for (int j = 0; j < 8; ++j) {
    int row = rbase + j;
    if (row < 15500) {
      float v0 = acc0[j] + bphi[n0 + r];
      float v1 = acc1[j] + bphi[n0 + 16 + r];
      XV[(long)row * HD + n0 + r]      = (__bf16)(v0 > 0.f ? v0 : 0.f);
      XV[(long)row * HD + n0 + 16 + r] = (__bf16)(v1 > 0.f ? v1 : 0.f);
    }
  }
}

// ---------------------------------------------------------------------------
// k_gi0: GI0[t,30,768] = x_t @ Wih0^T + bih0  (fp32 out)
// x_t[m, k<256] = xv[t,1+m,k] ; x_t[m, k>=256] = xv[t,0,k-256]
// ---------------------------------------------------------------------------
__global__ void k_gi0(const __bf16* __restrict__ XV, const __bf16* __restrict__ Wih0B,
                      const float* __restrict__ bih0, float* __restrict__ GI0) {
  const int lane = threadIdx.x & 31;
  const int wid  = threadIdx.x >> 5;
  const int tile = blockIdx.x * 8 + wid;              // 500*96 = 48000 tiles
  const int t  = tile / 96;
  const int rr = tile % 96;
  const int tm = rr / 48, tn = rr % 48;
  const int r  = lane & 15;
  const int kA = (lane & 16) ? 8 : 0;
  int m = tm * 16 + r; int mc = (m < NBX) ? m : (NBX - 1);
  const __bf16* xb  = XV + (long)t * 31 * HD;
  const __bf16* aLo = xb + (long)(1 + mc) * HD;   // xv[1+m]
  const __bf16* aHi = xb;                         // xv[0]
  const __bf16* bp  = Wih0B + (long)(tn * 16 + r) * 512 + kA;
  v8f acc = {};
  for (int kk = 0; kk < 512; kk += 32) {
    int k0 = kk + kA, k1 = kk + kA + 16;          // chunks never cross k=256
    const __bf16* p0 = (k0 < HD) ? (aLo + k0) : (aHi + (k0 - HD));
    const __bf16* p1 = (k1 < HD) ? (aLo + k1) : (aHi + (k1 - HD));
    v8bf lo = *(const v8bf*)p0, hi = *(const v8bf*)p1;
    v16bf a = __builtin_shufflevector(lo, hi, 0,1,2,3,4,5,6,7,8,9,10,11,12,13,14,15);
    v16bf b = ld_frag(bp + kk);
    acc = wmma_bf16(a, b, acc);
  }
  const int col = tn * 16 + r;
  const float bv = bih0[col];
  const int rbase = tm * 16 + ((lane & 16) ? 8 : 0);
#pragma unroll
  for (int j = 0; j < 8; ++j) {
    int row = rbase + j;
    if (row < NBX) GI0[((long)t * NBX + row) * 768 + col] = acc[j] + bv;
  }
}

// ---------------------------------------------------------------------------
// k_gic: GIc[t,30,96] = (cor/scale) @ Wc_ih^T + bc_ih  (K = 4, scalar)
// ---------------------------------------------------------------------------
__global__ void k_gic(const float* __restrict__ y, const float* __restrict__ Wc_ih,
                      const float* __restrict__ bc_ih, float* __restrict__ GIc) {
  int i = blockIdx.x * 256 + threadIdx.x;
  if (i >= TSTEPS * NBX * 96) return;
  int n = i % 96, m = (i / 96) % NBX, t = i / (96 * NBX);
  const float* yy = y + ((long)t * NBX + m) * 6;
  const float sc0 = 1080.f, sc1 = 720.f;
  float s = bc_ih[n];
  s += (yy[1] / sc0) * Wc_ih[n * 4 + 0];
  s += (yy[2] / sc1) * Wc_ih[n * 4 + 1];
  s += (yy[3] / sc0) * Wc_ih[n * 4 + 2];
  s += (yy[4] / sc1) * Wc_ih[n * 4 + 3];
  GIc[i] = s;
}

// ---------------------------------------------------------------------------
// Phase-2 LDS layout (dynamic shared; ~317 KB of the WGP's 320 KB)
// ---------------------------------------------------------------------------
constexpr int BF_ABF  = 0;                 // 32x288  h0 / feat A-staging
constexpr int BF_ABF2 = BF_ABF  + 32*288;  // 32x256  Hp0 / Hp1 A-staging
constexpr int BF_HCBF = BF_ABF2 + 32*256;  // 32x32   Hcp A-staging
constexpr int BF_TOT  = BF_HCBF + 32*32;

constexpr int F_HN0  = 0;                  // carry H layer0 (30x256)
constexpr int F_HN1  = F_HN0  + NBX*HD;    // carry H layer1
constexpr int F_HCN  = F_HN1  + NBX*HD;    // carry Hc (30x32)
constexpr int F_H0   = F_HCN  + NBX*HCD;   // h0 (30x256)
constexpr int F_H1   = F_H0   + NBX*HD;    // h1
constexpr int F_HC   = F_H1   + NBX*HD;    // hc (30x32)
constexpr int F_G    = F_HC   + NBX*HCD;   // gate buffer 32x768 (reused as L1)
constexpr int F_IN   = F_G    + 32*768;    // i_n slice layer1 (30x256)
constexpr int F_GC   = F_IN   + NBX*HD;    // hc gates (30x96)
constexpr int F_GICS = F_GC   + NBX*96;    // TDM-staged GIc[t] (30x96)
constexpr int F_WA   = F_GICS + NBX*96;    // Wa (256)
constexpr int F_WAC  = F_WA   + 256;       // Wac (32)
constexpr int F_W2   = F_WAC  + 32;        // W2 (2x256)
constexpr int F_SB   = F_W2   + 512;       // attn scores 3x32
constexpr int F_AL   = F_SB   + 96;        // attn alphas 3x32
constexpr int F_LG   = F_AL   + 96;        // logits 30x2 (+pad)
constexpr int F_MISC = F_LG   + 64;        // [0] loss accumulator
constexpr int F_TOT  = F_MISC + 4;
constexpr int I_TOT  = 128;                // actv/lab/pt/hp (30 each)
constexpr size_t SMEM_BYTES = (size_t)BF_TOT*2 + (size_t)F_TOT*4 + (size_t)I_TOT*4;

// gh = A(32x256 bf16, LDS) @ W(768x256 bf16)^T + bias -> G (30x768 f32, LDS)
__device__ __forceinline__ void gate_gemm(const __bf16* Asrc, const __bf16* Wg,
                                          const float* bias, float* Gout,
                                          int lane, int wid) {
  const int r  = lane & 15;
  const int kA = (lane & 16) ? 8 : 0;
  for (int tt = wid; tt < 96; tt += 16) {
    const int tm = tt / 48, tn = tt % 48;
    const int col = tn * 16 + r;
    const __bf16* ap = Asrc + (tm * 16 + r) * HD + kA;
    const __bf16* bp = Wg + (long)col * HD + kA;
    v8f acc; const float bv = bias[col];
#pragma unroll
    for (int j = 0; j < 8; ++j) acc[j] = bv;
    for (int kk = 0; kk < HD; kk += 32)
      acc = wmma_bf16(ld_frag(ap + kk), ld_frag(bp + kk), acc);
    const int rbase = tm * 16 + ((lane & 16) ? 8 : 0);
#pragma unroll
    for (int j = 0; j < 8; ++j) {
      int row = rbase + j;
      if (row < NBX) Gout[row * 768 + col] = acc[j];
    }
  }
}

// ---------------------------------------------------------------------------
// k_seq: 500-step recurrence, one workgroup of 512 threads (16 waves)
// ---------------------------------------------------------------------------
__global__ void __launch_bounds__(512)
k_seq(const float* __restrict__ GI0, const float* __restrict__ GIc,
      const __bf16* __restrict__ Whh0B, const __bf16* __restrict__ Wih1B,
      const __bf16* __restrict__ Whh1B, const __bf16* __restrict__ W1B,
      const __bf16* __restrict__ WchhB,
      const float* __restrict__ bhh0, const float* __restrict__ bih1,
      const float* __restrict__ bhh1, const float* __restrict__ b1,
      const float* __restrict__ W2, const float* __restrict__ b2,
      const float* __restrict__ Wa, const float* __restrict__ Wac,
      const float* __restrict__ bc_hh,
      const float* __restrict__ y, const int* __restrict__ prev_slot,
      const unsigned char* __restrict__ has_prev, float* __restrict__ out) {
  extern __shared__ __align__(16) char smem[];
  __bf16* bfb  = (__bf16*)smem;
  float*  fb   = (float*)(smem + (size_t)BF_TOT * 2);
  int*    ib   = (int*)(smem + (size_t)BF_TOT * 2 + (size_t)F_TOT * 4);

  __bf16* Abf  = bfb + BF_ABF;   // ld 288
  __bf16* Abf2 = bfb + BF_ABF2;  // ld 256
  __bf16* Hcbf = bfb + BF_HCBF;  // ld 32
  float *Hn0 = fb+F_HN0, *Hn1 = fb+F_HN1, *Hcn = fb+F_HCN;
  float *h0f = fb+F_H0,  *h1f = fb+F_H1,  *hcf = fb+F_HC;
  float *G   = fb+F_G,   *In  = fb+F_IN,  *Gc  = fb+F_GC;
  float *GICS= fb+F_GICS;
  float *WaL = fb+F_WA,  *WacL= fb+F_WAC, *W2L = fb+F_W2;
  float *sb  = fb+F_SB,  *al  = fb+F_AL,  *lg  = fb+F_LG, *misc = fb+F_MISC;
  int *actv = ib, *labL = ib+30, *ptL = ib+60, *hpL = ib+90;
  const unsigned gics_off = (unsigned)(size_t)(void*)GICS;  // LDS byte offset

  const int tid  = threadIdx.x;
  const int lane = tid & 31;
  const int wid  = tid >> 5;
  const int NT   = 512;
  const int r    = lane & 15;
  const int kA   = (lane & 16) ? 8 : 0;
  const int rsel = (lane & 16) ? 8 : 0;

  for (int i = tid; i < NBX*HD; i += NT) { Hn0[i] = 0.f; Hn1[i] = 0.f; }
  for (int i = tid; i < NBX*HCD; i += NT) Hcn[i] = 0.f;
  for (int i = tid; i < 256; i += NT) WaL[i] = Wa[i];
  for (int i = tid; i < 32;  i += NT) WacL[i] = Wac[i];
  for (int i = tid; i < 512; i += NT) W2L[i] = W2[i];
  if (tid == 0) misc[0] = 0.f;
  __syncthreads();

  for (int t = 0; t < TSTEPS; ++t) {
    // ---- TDM: stage GIc[t] (30x96 f32) into LDS, overlapped with stage 2 --
    if (wid == 15) tdm_load_1d_f32(GIc + (long)t * NBX * 96, gics_off, NBX * 96);

    // ---- per-step scalars -------------------------------------------------
    if (tid < NBX) {
      const float* yy = y + ((long)t * NBX + tid) * 6;
      actv[tid] = (yy[0] != 0.f) ? 1 : 0;
      labL[tid] = (int)yy[5];
      ptL[tid]  = prev_slot[t * NBX + tid];
      hpL[tid]  = has_prev[t * NBX + tid] ? 1 : 0;
    }
    __syncthreads();

    // ---- stage 1: gather Hp0 -> Abf2 (bf16), Hcp -> Hcbf ------------------
    for (int i = tid; i < 32 * HD; i += NT) {
      int m = i >> 8, k = i & 255;
      float v = (m < NBX && hpL[m]) ? Hn0[ptL[m] * HD + k] : 0.f;
      Abf2[i] = (__bf16)v;
    }
    for (int i = tid; i < 32 * HCD; i += NT) {
      int m = i >> 5, c = i & 31;
      float v = (m < NBX && hpL[m]) ? Hcn[ptL[m] * HCD + c] : 0.f;
      Hcbf[i] = (__bf16)v;
    }
    __syncthreads();

    // prefetch next-stage weights into L2/WGP$ (global_prefetch_b8)
    __builtin_prefetch(Whh1B + ((long)tid * 384), 0, 0);
    __builtin_prefetch(Wih1B + ((long)tid * 384), 0, 0);

    // ---- stage 2: gh0 = Hp0 @ Whh0^T + bhh0 -> G; ghc -> Gc ---------------
    gate_gemm(Abf2, Whh0B, bhh0, G, lane, wid);
    for (int tt = wid; tt < 12; tt += 16) {       // 30x96, K=32: one WMMA
      const int tm = tt / 6, tn = tt % 6;
      const int col = tn * 16 + r;
      const __bf16* ap = Hcbf + (tm * 16 + r) * HCD + kA;
      const __bf16* bp = WchhB + (long)col * HCD + kA;
      v8f acc; const float bv = bc_hh[col];
#pragma unroll
      for (int j = 0; j < 8; ++j) acc[j] = bv;
      acc = wmma_bf16(ld_frag(ap), ld_frag(bp), acc);
      const int rbase = tm * 16 + rsel;
#pragma unroll
      for (int j = 0; j < 8; ++j) {
        int row = rbase + j;
        if (row < NBX) Gc[row * 96 + col] = acc[j];
      }
    }
    if (wid == 15) __builtin_amdgcn_s_wait_tensorcnt((short)0);  // GICS ready
    __syncthreads();

    // ---- stage 3: GRU0 elementwise -> h0f, Abf=h0bf, Abf2=Hp1bf; hc GRU ---
    const float* gi0t = GI0 + (long)t * NBX * 768;
    for (int i = tid; i < 32 * HD; i += NT) {
      int m = i >> 8, k = i & 255;
      float h0v = 0.f;
      if (m < NBX) {
        float rr_ = sigm(gi0t[m*768 + k]       + G[m*768 + k]);
        float zz  = sigm(gi0t[m*768 + 256 + k] + G[m*768 + 256 + k]);
        float nn  = tanhf(gi0t[m*768 + 512 + k] + rr_ * G[m*768 + 512 + k]);
        float hp  = hpL[m] ? Hn0[ptL[m] * HD + k] : 0.f;
        h0v = (1.f - zz) * nn + zz * hp;
        h0f[m * HD + k] = h0v;
      }
      Abf[m * 288 + k] = (__bf16)h0v;
      float hp1 = (m < NBX && hpL[m]) ? Hn1[ptL[m] * HD + k] : 0.f;
      Abf2[i] = (__bf16)hp1;
    }
    for (int i = tid; i < NBX * HCD; i += NT) {   // hc GRU (GIc from TDM LDS)
      int m = i >> 5, c = i & 31;
      float rr_ = sigm(GICS[m*96 + c]      + Gc[m*96 + c]);
      float zz  = sigm(GICS[m*96 + 32 + c] + Gc[m*96 + 32 + c]);
      float nn  = tanhf(GICS[m*96 + 64 + c] + rr_ * Gc[m*96 + 64 + c]);
      float hp  = hpL[m] ? Hcn[ptL[m] * HCD + c] : 0.f;
      hcf[i] = (1.f - zz) * nn + zz * hp;
    }
    __syncthreads();

    // ---- stage 4: gh1 = Hp1 @ Whh1^T + bhh1 -> G --------------------------
    gate_gemm(Abf2, Whh1B, bhh1, G, lane, wid);
    __syncthreads();

    // ---- stage 5: gi1 = h0 @ Wih1^T + bih1; r/z accumulate onto G(C-op),
    //      i_n slice -> In -------------------------------------------------
    __builtin_prefetch(W1B + ((long)tid * 144), 0, 0);
    for (int tt = wid; tt < 96; tt += 16) {
      const int tm = tt / 48, tn = tt % 48;
      const int col = tn * 16 + r;
      const int rbase = tm * 16 + rsel;
      v8f acc;
      if (tn < 32) {                 // C := gh1 (rows 0..31 of G are valid mem)
#pragma unroll
        for (int j = 0; j < 8; ++j) acc[j] = G[(rbase + j) * 768 + col];
      } else {
#pragma unroll
        for (int j = 0; j < 8; ++j) acc[j] = 0.f;
      }
      const __bf16* ap = Abf + (tm * 16 + r) * 288 + kA;     // h0 bf16
      const __bf16* bp = Wih1B + (long)col * HD + kA;
      for (int kk = 0; kk < HD; kk += 32)
        acc = wmma_bf16(ld_frag(ap + kk), ld_frag(bp + kk), acc);
      const float bv = bih1[col];
#pragma unroll
      for (int j = 0; j < 8; ++j) {
        int row = rbase + j;
        if (row < NBX) {
          float v = acc[j] + bv;
          if (tn < 32) G[row * 768 + col] = v;               // i_r+h_r / i_z+h_z
          else         In[row * HD + (col - 512)] = v;       // i_n
        }
      }
    }
    __syncthreads();

    // ---- stage 6: GRU1 elementwise -> h1f; Abf = [h1 | hc] bf16 -----------
    for (int i = tid; i < 32 * 288; i += NT) {
      int m = i / 288, k = i % 288;
      float v = 0.f;
      if (k < HD) {
        if (m < NBX) {
          float rr_ = sigm(G[m*768 + k]);
          float zz  = sigm(G[m*768 + 256 + k]);
          float nn  = tanhf(In[m*HD + k] + rr_ * G[m*768 + 512 + k]);
          float hp  = hpL[m] ? Hn1[ptL[m] * HD + k] : 0.f;
          v = (1.f - zz) * nn + zz * hp;
          h1f[m * HD + k] = v;
        }
      } else if (m < NBX) {
        v = hcf[m * HCD + (k - HD)];
      }
      Abf[i] = (__bf16)v;
    }
    __syncthreads();

    // ---- stage 7: L1 = relu(feat @ W1^T + b1) -> G (ld 256), K = 288 ------
    for (int tt = wid; tt < 32; tt += 16) {
      const int tm = tt >> 4, tn = tt & 15;
      const int col = tn * 16 + r;
      const __bf16* ap = Abf + (tm * 16 + r) * 288 + kA;
      const __bf16* bp = W1B + (long)col * 288 + kA;
      v8f acc = {};
      for (int kk = 0; kk < 288; kk += 32)
        acc = wmma_bf16(ld_frag(ap + kk), ld_frag(bp + kk), acc);
      const float bv = b1[col];
      const int rbase = tm * 16 + rsel;
#pragma unroll
      for (int j = 0; j < 8; ++j) {
        int row = rbase + j;
        if (row < NBX) { float v = acc[j] + bv; G[row * HD + col] = v > 0.f ? v : 0.f; }
      }
    }
    __syncthreads();

    // ---- stage 8: logits, outs, NLL loss ----------------------------------
    if (tid < 60) {
      int m = tid >> 1, c = tid & 1;
      float s = b2[c];
      const float* w = W2L + c * HD;
      for (int j = 0; j < HD; ++j) s += G[m * HD + j] * w[j];
      lg[m * 2 + c] = s;
      out[1 + (long)t * 60 + m * 2 + c] = actv[m] ? s : 0.f;
    }
    __syncthreads();
    if (tid == 0) {
      float a = 0.f;
      for (int m = 0; m < NBX; ++m) if (actv[m]) {
        float l0 = lg[m*2], l1 = lg[m*2+1];
        float mx = l0 > l1 ? l0 : l1;
        float lse = mx + __logf(__expf(l0 - mx) + __expf(l1 - mx));
        a += lse - (labL[m] ? l1 : l0);
      }
      misc[0] += a;
    }
    // ---- stage 9: attention scores (waves 0/1/2, lane = object) -----------
    if (wid == 0 && lane < NBX) {
      float s = 0.f; const float* h = h0f + lane * HD;
      for (int k = 0; k < HD; ++k) s += tanhf(h[k]) * WaL[k];
      sb[lane] = s;
    } else if (wid == 1 && lane < NBX) {
      float s = 0.f; const float* h = h1f + lane * HD;
      for (int k = 0; k < HD; ++k) s += tanhf(h[k]) * WaL[k];
      sb[32 + lane] = s;
    } else if (wid == 2 && lane < NBX) {
      float s = 0.f; const float* h = hcf + lane * HCD;
      for (int k = 0; k < HCD; ++k) s += tanhf(h[k]) * WacL[k];
      sb[64 + lane] = s;
    }
    __syncthreads();
    if (tid < 3) {                          // masked softmax over objects
      const float* s = sb + tid * 32; float* a = al + tid * 32;
      float mx = -1e30f;
      for (int m = 0; m < NBX; ++m) if (actv[m] && s[m] > mx) mx = s[m];
      float sum = 0.f;
      for (int m = 0; m < NBX; ++m) { float e = actv[m] ? __expf(s[m]-mx) : 0.f; a[m] = e; sum += e; }
      float inv = (sum > 0.f) ? 1.f / sum : 0.f;
      for (int m = 0; m < NBX; ++m) a[m] *= inv;
    }
    __syncthreads();
    // ---- stage 10: carry update ------------------------------------------
    for (int i = tid; i < NBX * HD; i += NT) {
      int m = i >> 8;
      Hn0[i] = h0f[i] * al[m];
      Hn1[i] = h1f[i] * al[32 + m];
    }
    for (int i = tid; i < NBX * HCD; i += NT) {
      int m = i >> 5;
      Hcn[i] = hcf[i] * al[64 + m];
    }
    __syncthreads();
  }
  if (tid == 0) out[0] = misc[0];
}

// ---------------------------------------------------------------------------
extern "C" void kernel_launch(void* const* d_in, const int* in_sizes, int n_in,
                              void* d_out, int out_size, void* d_ws, size_t ws_size,
                              hipStream_t stream) {
  const float* flow  = (const float*)d_in[0];
  const float* y     = (const float*)d_in[1];
  const int*   pslot = (const int*)d_in[2];
  const unsigned char* hprev = (const unsigned char*)d_in[3];
  const float* Wphi = (const float*)d_in[4];
  const float* bphi = (const float*)d_in[5];
  const float* Wih0 = (const float*)d_in[6];
  const float* Whh0 = (const float*)d_in[7];
  const float* bih0 = (const float*)d_in[8];
  const float* bhh0 = (const float*)d_in[9];
  const float* Wih1 = (const float*)d_in[10];
  const float* Whh1 = (const float*)d_in[11];
  const float* bih1 = (const float*)d_in[12];
  const float* bhh1 = (const float*)d_in[13];
  const float* W1   = (const float*)d_in[14];
  const float* b1   = (const float*)d_in[15];
  const float* W2   = (const float*)d_in[16];
  const float* b2   = (const float*)d_in[17];
  const float* Wa   = (const float*)d_in[18];
  const float* Wcih = (const float*)d_in[19];
  const float* Wchh = (const float*)d_in[20];
  const float* bcih = (const float*)d_in[21];
  const float* bchh = (const float*)d_in[22];
  const float* Wac  = (const float*)d_in[23];

  char* ws = (char*)d_ws;
  size_t off = 0;
  auto carve = [&](size_t bytes) { void* p = ws + off; off = (off + bytes + 255) & ~(size_t)255; return p; };
  __bf16* WphiB = (__bf16*)carve(256*2048*2);
  __bf16* Wih0B = (__bf16*)carve(768*512*2);
  __bf16* Whh0B = (__bf16*)carve(768*256*2);
  __bf16* Wih1B = (__bf16*)carve(768*256*2);
  __bf16* Whh1B = (__bf16*)carve(768*256*2);
  __bf16* W1B   = (__bf16*)carve(256*288*2);
  __bf16* WchhB = (__bf16*)carve(96*32*2);
  __bf16* XV    = (__bf16*)carve((size_t)15500*256*2);
  float*  GI0   = (float*)carve((size_t)500*30*768*4);
  float*  GIc   = (float*)carve((size_t)500*30*96*4);

  // weight conversion fp32 -> bf16
  k_f2bf<<<(256*2048+255)/256, 256, 0, stream>>>(Wphi, WphiB, 256*2048);
  k_f2bf<<<(768*512 +255)/256, 256, 0, stream>>>(Wih0, Wih0B, 768*512);
  k_f2bf<<<(768*256 +255)/256, 256, 0, stream>>>(Whh0, Whh0B, 768*256);
  k_f2bf<<<(768*256 +255)/256, 256, 0, stream>>>(Wih1, Wih1B, 768*256);
  k_f2bf<<<(768*256 +255)/256, 256, 0, stream>>>(Whh1, Whh1B, 768*256);
  k_f2bf<<<(256*288 +255)/256, 256, 0, stream>>>(W1,   W1B,   256*288);
  k_f2bf<<<(96*32   +255)/256, 256, 0, stream>>>(Wchh, WchhB, 96*32);

  // phase 1: batched GEMMs (WMMA bf16)
  k_xv <<<969, 256, 0, stream>>>(flow, WphiB, bphi, XV);        // 16-row blocks
  k_gi0<<<6000, 256, 0, stream>>>(XV, Wih0B, bih0, GI0);        // 48000 tiles
  k_gic<<<(500*30*96+255)/256, 256, 0, stream>>>(y, Wcih, bcih, GIc);

  // phase 2: persistent recurrence (1 WGP, LDS-resident state, TDM staging)
  k_seq<<<1, 512, SMEM_BYTES, stream>>>(GI0, GIc, Whh0B, Wih1B, Whh1B, W1B, WchhB,
                                        bhh0, bih1, bhh1, b1, W2, b2, Wa, Wac, bchh,
                                        y, pslot, hprev, (float*)d_out);
}
